// attention_49314814492995
// MI455X (gfx1250) — compile-verified, matrix operand order checked by reading, not run on hardware
//
#include <hip/hip_runtime.h>

// Problem dims
#define B_    128
#define D_    256
#define T_    32
#define C_    768
#define HW_   289          // 17*17
#define HWP_  320          // HW padded to K-multiple of 64
#define M_    (B_ * HW_)   // 36992 flat rows, /16 = 2312 exact

typedef __attribute__((ext_vector_type(16))) __bf16 v16bf;
typedef __attribute__((ext_vector_type(8)))  __bf16 v8bf;
typedef __attribute__((ext_vector_type(4)))  __bf16 v4bf;
typedef __attribute__((ext_vector_type(8)))  float  v8f;

__device__ __forceinline__ __bf16 cvt_bf16(float x) { return (__bf16)x; }

// ---------------------------------------------------------------------------
// bf16 fragment loader, contiguous-K layout: element(r,k) = base[r*rstride + k]
// Per lane: two 16-byte loads (8 bf16 each) -> global_load_b128.
// ---------------------------------------------------------------------------
__device__ __forceinline__ v16bf load_frag16(const __bf16* __restrict__ base,
                                             int rstride, int r0, int k0) {
  const int lane = threadIdx.x & 31;
  const __bf16* p = base + (size_t)(r0 + (lane & 15)) * rstride
                         + k0 + ((lane >> 4) << 3);
  v8bf lo = *(const v8bf*)(p);        // K offsets +0..7
  v8bf hi = *(const v8bf*)(p + 16);   // K offsets +16..23
  return __builtin_shufflevector(lo, hi, 0,1,2,3,4,5,6,7,8,9,10,11,12,13,14,15);
}

__device__ __forceinline__ v8f wmma_bf16(v16bf a, v16bf b, v8f acc) {
  return __builtin_amdgcn_wmma_f32_16x16x32_bf16(false, a, false, b,
                                                 (short)0, acc, false, false);
}

// ---------------------------------------------------------------------------
// GEMM strip: one A tile row-strip vs JT B tiles. Single-buffered slice body
// (low register pressure), loads grouped ahead of the WMMAs; `unroll 2` lets
// the scheduler overlap slice k+32 loads with slice k WMMAs without spills.
// ---------------------------------------------------------------------------
template <int KTOT, int JT>
__device__ __forceinline__ void gemm_strip(const __bf16* __restrict__ A, int lda, int r0,
                                           const __bf16* __restrict__ Bm, int ldb, int c0,
                                           v8f (&acc)[JT]) {
#pragma unroll 2
  for (int k = 0; k < KTOT; k += 32) {
    __builtin_prefetch(A + (size_t)r0 * lda + k + 128, 0, 3);
    const v16bf a = load_frag16(A, lda, r0, k);
    v16bf b[JT];
#pragma unroll
    for (int j = 0; j < JT; ++j) b[j] = load_frag16(Bm, ldb, c0 + j * 16, k);
#pragma unroll
    for (int j = 0; j < JT; ++j) acc[j] = wmma_bf16(a, b[j], acc[j]);
  }
}

// ---------------------------------------------------------------------------
// Pack kernels (one-pass, bandwidth bound)
// ---------------------------------------------------------------------------
__global__ void k_pack_f(const float* __restrict__ f, __bf16* __restrict__ fbf) {
  const size_t i = (size_t)blockIdx.x * blockDim.x + threadIdx.x;
  if (i >= (size_t)M_ * C_ / 4) return;
  const float4 v = ((const float4*)f)[i];
  v4bf o;
  o[0] = cvt_bf16(v.x); o[1] = cvt_bf16(v.y);
  o[2] = cvt_bf16(v.z); o[3] = cvt_bf16(v.w);
  ((v4bf*)fbf)[i] = o;
}

// W fp32 [C][D] -> Wt bf16 [D][C]
__global__ void k_pack_w(const float* __restrict__ W, __bf16* __restrict__ Wt) {
  const int i = blockIdx.x * blockDim.x + threadIdx.x;
  if (i >= C_ * D_) return;
  const int d = i / C_, c = i % C_;
  Wt[i] = cvt_bf16(W[(size_t)c * D_ + d]);
}

// e fp32 [B][D][T] -> ebt bf16 [B][T][D]
__global__ void k_pack_e(const float* __restrict__ e, __bf16* __restrict__ ebt) {
  const int i = blockIdx.x * blockDim.x + threadIdx.x;
  if (i >= B_ * T_ * D_) return;
  const int d = i % D_, t = (i / D_) % T_, b = i / (T_ * D_);
  ebt[i] = cvt_bf16(e[((size_t)b * D_ + d) * T_ + t]);
}

// zero the K-pad of v1: v1[row][289..319], row in [0, B*D)
__global__ void k_zero_v1pad(__bf16* __restrict__ v1) {
  const int i = blockIdx.x * blockDim.x + threadIdx.x;
  if (i >= B_ * D_ * (HWP_ - HW_)) return;
  const int row = i / (HWP_ - HW_), off = i % (HWP_ - HW_);
  v1[(size_t)row * HWP_ + HW_ + off] = cvt_bf16(0.0f);
}

// ---------------------------------------------------------------------------
// K1: projection GEMM  v = f @ W + b   (M=36992, N=256, K=768)
// One wave per (m-tile, d-quarter-strip of 4 tiles).
// Writes v in both bf16 layouts: v1[b][d][nP] and vt2[b][nP][d].
// ---------------------------------------------------------------------------
__global__ void __launch_bounds__(256)
k_proj_gemm(const __bf16* __restrict__ fbf,
            const __bf16* __restrict__ Wt,
            const float*  __restrict__ bias,
            __bf16* __restrict__ v1, __bf16* __restrict__ vt2) {
  const int wave = (int)((blockIdx.x * blockDim.x + threadIdx.x) >> 5);
  if (wave >= (M_ / 16) * 4) return;
  const int mt = wave >> 2;           // 0..2311
  const int nh = wave & 3;            // which 4-tile quarter of D
  const int m0 = mt * 16;
  const int c0 = nh * 64;

  v8f acc[4] = {};
  gemm_strip<C_, 4>(fbf, C_, m0, Wt, C_, c0, acc);

  const int lane = threadIdx.x & 31;
  const int half = lane >> 4, nq = lane & 15;
  int bb[8], ns[8];
#pragma unroll
  for (int r = 0; r < 8; ++r) {       // hoisted: 8 divisions per wave total
    const int m = m0 + r + half * 8;
    bb[r] = m / HW_;
    ns[r] = m - bb[r] * HW_;
  }
#pragma unroll
  for (int j = 0; j < 4; ++j) {
    const int d = c0 + j * 16 + nq;
    const float bv = bias[d];
#pragma unroll
    for (int r = 0; r < 8; ++r) {
      const __bf16 h = cvt_bf16(acc[j][r] + bv);
      v1 [((size_t)bb[r] * D_   + d)     * HWP_ + ns[r]] = h;
      vt2[((size_t)bb[r] * HWP_ + ns[r]) * D_   + d]     = h;
    }
  }
}

// ---------------------------------------------------------------------------
// K2: s[b][t][n] = sum_d e[b][d][t] v[b][d][n]  (M=32, N=289, K=256)
// ---------------------------------------------------------------------------
__global__ void __launch_bounds__(256)
k_s_gemm(const __bf16* __restrict__ ebt,
         const __bf16* __restrict__ vt2,
         float* __restrict__ s) {
  const int NT = 19, TT = 2;
  const int wave = (int)((blockIdx.x * blockDim.x + threadIdx.x) >> 5);
  if (wave >= B_ * TT * NT) return;
  const int b  = wave / (TT * NT);
  const int tt = (wave / NT) % TT;
  const int nt = wave % NT;
  const int m0 = tt * 16, n0 = nt * 16;

  const __bf16* A  = ebt + (size_t)b * T_   * D_;
  const __bf16* Bm = vt2 + (size_t)b * HWP_ * D_;
  v8f acc[1] = {};
  gemm_strip<D_, 1>(A, D_, m0, Bm, D_, n0, acc);

  const int lane = threadIdx.x & 31;
  const int half = lane >> 4;
  const int n = n0 + (lane & 15);
  if (n < HW_) {
#pragma unroll
    for (int r = 0; r < 8; ++r) {
      const int t = m0 + r + half * 8;
      s[((size_t)b * T_ + t) * HW_ + n] = acc[0][r];
    }
  }
}

// ---------------------------------------------------------------------------
// K3: softmax over T per (b,n); in-place s <- gamma * softmax_T(s).
// T = 32 = wave32: one wave per column.
// ---------------------------------------------------------------------------
__global__ void k_softmax_T(float* __restrict__ s, const int* __restrict__ gamma_p) {
  const int wave = (int)((blockIdx.x * blockDim.x + threadIdx.x) >> 5);
  if (wave >= B_ * HW_) return;
  const int b = wave / HW_, n = wave % HW_;
  const int t = threadIdx.x & 31;
  float* p = s + ((size_t)b * T_ + t) * HW_ + n;
  const float x = *p;
  float mx = x;
#pragma unroll
  for (int o = 16; o > 0; o >>= 1) mx = fmaxf(mx, __shfl_xor(mx, o, 32));
  const float ex = __expf(x - mx);
  float sm = ex;
#pragma unroll
  for (int o = 16; o > 0; o >>= 1) sm += __shfl_xor(sm, o, 32);
  *p = (float)gamma_p[0] * (ex / sm);
}

// ---------------------------------------------------------------------------
// K4: alpha[b][t][:] = softmax over HW; emitted as bf16, K-padded with zeros.
// ---------------------------------------------------------------------------
__global__ void k_softmax_HW(const float* __restrict__ s, __bf16* __restrict__ alpha) {
  const int wave = (int)((blockIdx.x * blockDim.x + threadIdx.x) >> 5);
  if (wave >= B_ * T_) return;
  const int b = wave / T_, t = wave % T_;
  const int lane = threadIdx.x & 31;
  const float* p = s     + ((size_t)b * T_ + t) * HW_;
  __bf16*      q = alpha + ((size_t)b * T_ + t) * HWP_;

  float mx = -3.0e38f;
  for (int n = lane; n < HW_; n += 32) mx = fmaxf(mx, p[n]);
#pragma unroll
  for (int o = 16; o > 0; o >>= 1) mx = fmaxf(mx, __shfl_xor(mx, o, 32));
  float sm = 0.0f;
  for (int n = lane; n < HW_; n += 32) sm += __expf(p[n] - mx);
#pragma unroll
  for (int o = 16; o > 0; o >>= 1) sm += __shfl_xor(sm, o, 32);
  const float inv = 1.0f / sm;
  for (int n = lane; n < HWP_; n += 32)
    q[n] = cvt_bf16(n < HW_ ? __expf(p[n] - mx) * inv : 0.0f);
}

// ---------------------------------------------------------------------------
// K5: c[b][d][t] = sum_n v[b][d][n] alpha[b][t][n]  (M=256, N=32, K=320 padded)
// ---------------------------------------------------------------------------
__global__ void __launch_bounds__(256)
k_c_gemm(const __bf16* __restrict__ v1,
         const __bf16* __restrict__ alpha,
         float* __restrict__ c) {
  const int MT = D_ / 16, NT = T_ / 16;   // 16, 2
  const int wave = (int)((blockIdx.x * blockDim.x + threadIdx.x) >> 5);
  if (wave >= B_ * MT * NT) return;
  const int b  = wave / (MT * NT);
  const int mt = (wave / NT) % MT;
  const int nt = wave % NT;
  const int m0 = mt * 16, n0 = nt * 16;

  const __bf16* A  = v1    + (size_t)b * D_ * HWP_;
  const __bf16* Bm = alpha + (size_t)b * T_ * HWP_;
  v8f acc[1] = {};
  gemm_strip<HWP_, 1>(A, HWP_, m0, Bm, HWP_, n0, acc);

  const int lane = threadIdx.x & 31;
  const int half = lane >> 4;
  const int t = n0 + (lane & 15);
#pragma unroll
  for (int r = 0; r < 8; ++r) {
    const int d = m0 + r + half * 8;
    c[((size_t)b * D_ + d) * T_ + t] = acc[0][r];
  }
}

// ---------------------------------------------------------------------------
// K6: inverse L2 norms over D for c and e, one wave per (b,t).
// ---------------------------------------------------------------------------
__global__ void k_norms(const float* __restrict__ c, const float* __restrict__ e,
                        float* __restrict__ ilc, float* __restrict__ ile) {
  const int wave = (int)((blockIdx.x * blockDim.x + threadIdx.x) >> 5);
  if (wave >= B_ * T_) return;
  const int b = wave / T_, t = wave % T_;
  const int lane = threadIdx.x & 31;
  float sc = 0.0f, se = 0.0f;
  for (int d = lane; d < D_; d += 32) {
    const size_t idx = ((size_t)b * D_ + d) * T_ + t;
    const float cv = c[idx]; sc += cv * cv;
    const float ev = e[idx]; se += ev * ev;
  }
#pragma unroll
  for (int o = 16; o > 0; o >>= 1) {
    sc += __shfl_xor(sc, o, 32);
    se += __shfl_xor(se, o, 32);
  }
  if (lane == 0) {
    ilc[b * T_ + t] = __frsqrt_rn(sc);
    ile[b * T_ + t] = __frsqrt_rn(se);
  }
}

// ---------------------------------------------------------------------------
// K7: normalize + pack: cn[t][i][d], en[t][j][d] bf16 (contiguous along K=d)
// ---------------------------------------------------------------------------
__global__ void k_pack_cn_en(const float* __restrict__ c, const float* __restrict__ e,
                             const float* __restrict__ ilc, const float* __restrict__ ile,
                             __bf16* __restrict__ cn, __bf16* __restrict__ en) {
  const int wave = (int)((blockIdx.x * blockDim.x + threadIdx.x) >> 5);
  if (wave >= B_ * T_) return;
  const int i = wave / T_, t = wave % T_;
  const int lane = threadIdx.x & 31;
  const float sc = ilc[i * T_ + t];
  const float se = ile[i * T_ + t];
  for (int d = lane; d < D_; d += 32) {
    const size_t src = ((size_t)i * D_ + d) * T_ + t;
    const size_t dst = ((size_t)t * B_ + i) * D_ + d;
    cn[dst] = cvt_bf16(c[src] * sc);
    en[dst] = cvt_bf16(e[src] * se);
  }
}

// ---------------------------------------------------------------------------
// K8: cosine[i][j][t] = sum_d cn[t][i][d] en[t][j][d]  (per-t 128x128, K=256)
// One wave per (t, i-tile, j-half): 4-tile j-strip.
// ---------------------------------------------------------------------------
__global__ void __launch_bounds__(256)
k_cosine(const __bf16* __restrict__ cn, const __bf16* __restrict__ en,
         float* __restrict__ out) {
  const int IT = B_ / 16;                  // 8
  const int wave = (int)((blockIdx.x * blockDim.x + threadIdx.x) >> 5);
  if (wave >= T_ * IT * 2) return;
  const int t  = wave / (IT * 2);
  const int it = (wave / 2) % IT;
  const int jh = wave & 1;
  const int i0 = it * 16, j0 = jh * 64;

  const __bf16* A  = cn + (size_t)t * B_ * D_;
  const __bf16* Bm = en + (size_t)t * B_ * D_;
  v8f acc[4] = {};
  gemm_strip<D_, 4>(A, D_, i0, Bm, D_, j0, acc);

  const int lane = threadIdx.x & 31;
  const int half = lane >> 4, nq = lane & 15;
#pragma unroll
  for (int j = 0; j < 4; ++j) {
    const int jj = j0 + j * 16 + nq;
#pragma unroll
    for (int r = 0; r < 8; ++r) {
      const int i = i0 + r + half * 8;
      out[((size_t)i * B_ + jj) * T_ + t] = acc[j][r];
    }
  }
}

// ---------------------------------------------------------------------------
extern "C" void kernel_launch(void* const* d_in, const int* in_sizes, int n_in,
                              void* d_out, int out_size, void* d_ws, size_t ws_size,
                              hipStream_t stream) {
  const float* e     = (const float*)d_in[0];   // [B, D, T]
  const float* f     = (const float*)d_in[1];   // [B, H, W, C] = [M][C]
  const int*   gamma = (const int*)  d_in[2];
  const float* W     = (const float*)d_in[3];   // [C, D]
  const float* bias  = (const float*)d_in[4];   // [D]
  float* out = (float*)d_out;                   // [B, B, T]

  // Workspace carve-up
  char* w = (char*)d_ws;
  __bf16* fbf = (__bf16*)w;                 w += (size_t)M_ * C_ * 2;        // 56.8 MB
  __bf16* Wt  = (__bf16*)w;                 w += (size_t)D_ * C_ * 2;        //  0.4 MB
  __bf16* ebt = (__bf16*)w;                 w += (size_t)B_ * T_ * D_ * 2;   //  2.0 MB
  __bf16* v1  = (__bf16*)w;                 w += (size_t)B_ * D_ * HWP_ * 2; // 21.0 MB
  __bf16* vt2 = (__bf16*)w;                 w += (size_t)B_ * HWP_ * D_ * 2; // 21.0 MB
  float*  s   = (float*)w;                  w += (size_t)B_ * T_ * HW_ * 4;  //  4.7 MB
  __bf16* al  = (__bf16*)w;                 w += (size_t)B_ * T_ * HWP_ * 2; //  2.6 MB
  float*  c   = (float*)w;                  w += (size_t)B_ * D_ * T_ * 4;   //  4.2 MB
  float*  ilc = (float*)w;                  w += (size_t)B_ * T_ * 4;
  float*  ile = (float*)w;                  w += (size_t)B_ * T_ * 4;
  __bf16* cn  = (__bf16*)w;                 w += (size_t)T_ * B_ * D_ * 2;   //  2.0 MB
  __bf16* en  = (__bf16*)w;                 /* 2.0 MB */

  const int TPB = 256, WPB = TPB / 32;

  // Packing passes
  k_pack_f<<<((size_t)M_ * C_ / 4 + TPB - 1) / TPB, TPB, 0, stream>>>(f, fbf);
  k_pack_w<<<(C_ * D_ + TPB - 1) / TPB, TPB, 0, stream>>>(W, Wt);
  k_pack_e<<<(B_ * T_ * D_ + TPB - 1) / TPB, TPB, 0, stream>>>(e, ebt);
  k_zero_v1pad<<<(B_ * D_ * (HWP_ - HW_) + TPB - 1) / TPB, TPB, 0, stream>>>(v1);

  // Pipeline
  k_proj_gemm<<<((M_ / 16) * 4 + WPB - 1) / WPB, TPB, 0, stream>>>(fbf, Wt, bias, v1, vt2);
  k_s_gemm<<<(B_ * 2 * 19 + WPB - 1) / WPB, TPB, 0, stream>>>(ebt, vt2, s);
  k_softmax_T<<<(B_ * HW_ + WPB - 1) / WPB, TPB, 0, stream>>>(s, gamma);
  k_softmax_HW<<<(B_ * T_ + WPB - 1) / WPB, TPB, 0, stream>>>(s, al);
  k_c_gemm<<<(B_ * 16 * 2 + WPB - 1) / WPB, TPB, 0, stream>>>(v1, al, c);
  k_norms<<<(B_ * T_ + WPB - 1) / WPB, TPB, 0, stream>>>(c, e, ilc, ile);
  k_pack_cn_en<<<(B_ * T_ + WPB - 1) / WPB, TPB, 0, stream>>>(c, e, ilc, ile, cn, en);
  k_cosine<<<(T_ * 8 * 2 + WPB - 1) / WPB, TPB, 0, stream>>>(cn, en, out);
}
